// GAT_19808389169860
// MI455X (gfx1250) — matrix operand order repaired and checked by reading, not compiled, not run on hardware
//
#include <hip/hip_runtime.h>

#define N_NODES 100000
#define N_EDGES 1600000
#define HID 128
#define NCLS 10
#define NEG_SLOPE 0.2f

#define ROWS_PER_BLOCK 32
#define LDS_STRIDE 136   // 128 + 8 pad: 272B row stride -> conflict-free b128 frags

typedef __attribute__((ext_vector_type(16))) __bf16 v16bf;
typedef __attribute__((ext_vector_type(8)))  __bf16 v8bf;
typedef __attribute__((ext_vector_type(8)))  float  v8f;

// round-to-nearest-even f32 -> bf16 bits
static __device__ __forceinline__ unsigned bf_hi_bits(float f) {
  unsigned u = __float_as_uint(f);
  return (u + 0x7fffu + ((u >> 16) & 1u)) >> 16;
}

static __device__ __forceinline__ v16bf cat16(v8bf a, v8bf b) {
  return __builtin_shufflevector(a, b, 0,1,2,3,4,5,6,7,8,9,10,11,12,13,14,15);
}

// ---------------------------------------------------------------------------
// Weight preprocessing: W[k][c] (f32, row-major 128x128) -> transposed bf16
// hi/lo split WT[c][k] so B fragments are contiguous-in-K 16B loads.
// ---------------------------------------------------------------------------
__global__ __launch_bounds__(256) void wconv_kernel(const float* __restrict__ W,
                                                    unsigned short* __restrict__ WThi,
                                                    unsigned short* __restrict__ WTlo) {
  int i = blockIdx.x * blockDim.x + threadIdx.x;   // i = k*HID + c, 16384 threads
  int k = i >> 7, c = i & 127;
  float f = W[i];
  unsigned hb = bf_hi_bits(f);
  float lo = f - __uint_as_float(hb << 16);
  WThi[c * HID + k] = (unsigned short)hb;
  WTlo[c * HID + k] = (unsigned short)bf_hi_bits(lo);
}

// ---------------------------------------------------------------------------
// H = X @ W  (f32 in/out) via bf16x3 split WMMA, fp32 accumulate.
// Block = 256 threads = 8 waves; block computes 32 rows x 128 cols.
// Phase 1: cooperative f32->bf16 hi/lo conversion of the A tile into LDS
//          (done once per block instead of once per wave).
// Phase 2: wave w computes column tile w for 2 row tiles; A fragments come
//          from LDS as 16B ds loads; B fragments (preconverted weights) are
//          16B global loads reused across both row tiles.
// 24 v_wmma per wave. Grid exact (100000/32): EXEC all-1s for WMMA.
// ---------------------------------------------------------------------------
__global__ __launch_bounds__(256) void gemm_kernel(const float* __restrict__ X,
                                                   const unsigned short* __restrict__ WThi,
                                                   const unsigned short* __restrict__ WTlo,
                                                   float* __restrict__ H) {
  __shared__ __attribute__((aligned(16))) __bf16 lds_hi[ROWS_PER_BLOCK * LDS_STRIDE];
  __shared__ __attribute__((aligned(16))) __bf16 lds_lo[ROWS_PER_BLOCK * LDS_STRIDE];

  const int tid  = threadIdx.x;
  const int row0 = blockIdx.x * ROWS_PER_BLOCK;

  // ---- Phase 1: stage A tile (32 x 128) as bf16 hi/lo in LDS ----
  {
    const int r  = tid >> 3;            // 0..31
    const int k0 = (tid & 7) * 16;      // 0,16,...,112
    const float* __restrict__ src = X + (size_t)(row0 + r) * HID + k0;
    __bf16* dh = lds_hi + r * LDS_STRIDE + k0;
    __bf16* dl = lds_lo + r * LDS_STRIDE + k0;
#pragma unroll
    for (int j = 0; j < 16; ++j) {
      float f = src[j];
      unsigned hb = bf_hi_bits(f);
      float lo = f - __uint_as_float(hb << 16);
      dh[j] = __builtin_bit_cast(__bf16, (unsigned short)hb);
      dl[j] = __builtin_bit_cast(__bf16, (unsigned short)bf_hi_bits(lo));
    }
  }
  __syncthreads();

  // ---- Phase 2: WMMA ----
  const int lane = tid & 31;
  const int wave = tid >> 5;            // column tile 0..7
  const int half = lane >> 4;           // 0 or 1
  const int m    = lane & 15;
  const int col  = wave * 16 + m;

  const __bf16* __restrict__ bhiP = (const __bf16*)(WThi + (size_t)col * HID);
  const __bf16* __restrict__ bloP = (const __bf16*)(WTlo + (size_t)col * HID);
  const __bf16* a0h = lds_hi + m * LDS_STRIDE;          // row tile 0
  const __bf16* a0l = lds_lo + m * LDS_STRIDE;
  const __bf16* a1h = lds_hi + (m + 16) * LDS_STRIDE;   // row tile 1
  const __bf16* a1l = lds_lo + (m + 16) * LDS_STRIDE;

  v8f acc0 = {}, acc1 = {};
#pragma unroll
  for (int kk = 0; kk < 4; ++kk) {
    const int ka = kk * 32 + half * 8;   // A: K interleaved per half-wave
    const int kb = kk * 32 + half * 8;   // B: K interleaved per half-wave

    v16bf b_hi = cat16(*(const v8bf*)(bhiP + kb), *(const v8bf*)(bhiP + kb + 16));
    v16bf b_lo = cat16(*(const v8bf*)(bloP + kb), *(const v8bf*)(bloP + kb + 16));

    v16bf a0_hi = cat16(*(const v8bf*)(a0h + ka), *(const v8bf*)(a0h + ka + 16));
    v16bf a0_lo = cat16(*(const v8bf*)(a0l + ka), *(const v8bf*)(a0l + ka + 16));
    v16bf a1_hi = cat16(*(const v8bf*)(a1h + ka), *(const v8bf*)(a1h + ka + 16));
    v16bf a1_lo = cat16(*(const v8bf*)(a1l + ka), *(const v8bf*)(a1l + ka + 16));

    acc0 = __builtin_amdgcn_wmma_f32_16x16x32_bf16(false, a0_lo, false, b_hi, (short)0, acc0, false, false);
    acc0 = __builtin_amdgcn_wmma_f32_16x16x32_bf16(false, a0_hi, false, b_lo, (short)0, acc0, false, false);
    acc0 = __builtin_amdgcn_wmma_f32_16x16x32_bf16(false, a0_hi, false, b_hi, (short)0, acc0, false, false);
    acc1 = __builtin_amdgcn_wmma_f32_16x16x32_bf16(false, a1_lo, false, b_hi, (short)0, acc1, false, false);
    acc1 = __builtin_amdgcn_wmma_f32_16x16x32_bf16(false, a1_hi, false, b_lo, (short)0, acc1, false, false);
    acc1 = __builtin_amdgcn_wmma_f32_16x16x32_bf16(false, a1_hi, false, b_hi, (short)0, acc1, false, false);
  }

  const int mrow = row0 + half * 8;      // C/D layout: rows half*8 + i
#pragma unroll
  for (int i = 0; i < 8; ++i) {
    H[(size_t)(mrow + i)      * HID + col] = acc0[i];
    H[(size_t)(mrow + 16 + i) * HID + col] = acc1[i];
  }
}

// ---------------------------------------------------------------------------
// Per-node: alpha_s = h . a_src, alpha_d = h . a_dst; init max/denom/agg.
// One wave per node; grid exact (100000/8).
// ---------------------------------------------------------------------------
__global__ __launch_bounds__(256) void alpha_init_kernel(const float* __restrict__ H,
    const float* __restrict__ a_src, const float* __restrict__ a_dst,
    float* __restrict__ AS, float* __restrict__ AD,
    unsigned* __restrict__ MX, float* __restrict__ DEN,
    float* __restrict__ AGG) {
  const int lane = threadIdx.x & 31;
  const int node = blockIdx.x * 8 + (threadIdx.x >> 5);
  float4 hv = ((const float4*)(H + (size_t)node * HID))[lane];
  float4 sv = ((const float4*)a_src)[lane];
  float4 dv = ((const float4*)a_dst)[lane];
  float s = hv.x * sv.x + hv.y * sv.y + hv.z * sv.z + hv.w * sv.w;
  float d = hv.x * dv.x + hv.y * dv.y + hv.z * dv.z + hv.w * dv.w;
#pragma unroll
  for (int off = 16; off > 0; off >>= 1) {
    s += __shfl_down(s, off, 32);
    d += __shfl_down(d, off, 32);
  }
  if (lane == 0) { AS[node] = s; AD[node] = d; MX[node] = 0u; DEN[node] = 0.f; }
  float4 z = make_float4(0.f, 0.f, 0.f, 0.f);
  ((float4*)(AGG + (size_t)node * HID))[lane] = z;
}

// ---------------------------------------------------------------------------
// Edge pass 1: e = leaky_relu(as[src]+ad[dst]); store; segment-max via
// monotone-uint atomicMax (self-loops guarantee every dst is covered).
// ---------------------------------------------------------------------------
__global__ __launch_bounds__(256) void edge_max_kernel(const long long* __restrict__ ei,
    const float* __restrict__ AS, const float* __restrict__ AD,
    float* __restrict__ ELOG, unsigned* __restrict__ MX) {
  const int ET = N_EDGES + N_NODES;
  int e = blockIdx.x * blockDim.x + threadIdx.x;
  if (e >= ET) return;
  int s, d;
  if (e < N_EDGES) { s = (int)ei[e]; d = (int)ei[N_EDGES + e]; }
  else             { s = d = e - N_EDGES; }
  float v = AS[s] + AD[d];
  v = v > 0.f ? v : NEG_SLOPE * v;
  ELOG[e] = v;
  unsigned u = __float_as_uint(v);
  u = (u & 0x80000000u) ? ~u : (u | 0x80000000u);
  atomicMax(MX + d, u);
}

// ---------------------------------------------------------------------------
// Edge pass 2: ex = exp(e - m[dst]); store over ELOG; segment-sum denom.
// ---------------------------------------------------------------------------
__global__ __launch_bounds__(256) void edge_exp_kernel(const long long* __restrict__ ei,
    float* __restrict__ ELOG, const unsigned* __restrict__ MX,
    float* __restrict__ DEN) {
  const int ET = N_EDGES + N_NODES;
  int e = blockIdx.x * blockDim.x + threadIdx.x;
  if (e >= ET) return;
  int d = (e < N_EDGES) ? (int)ei[N_EDGES + e] : (e - N_EDGES);
  unsigned u = MX[d];
  unsigned b = (u & 0x80000000u) ? (u & 0x7fffffffu) : ~u;
  float ex = __expf(ELOG[e] - __uint_as_float(b));
  ELOG[e] = ex;
  atomicAdd(DEN + d, ex);
}

// ---------------------------------------------------------------------------
// Edge pass 3: agg[dst] += h[src] * (ex / (den[dst]+1e-16)).
// One wave per edge: 512B gather (L2-resident h) + 4 f32 atomics per lane.
// Grid exact (1700000/8).
// ---------------------------------------------------------------------------
__global__ __launch_bounds__(256) void aggregate_kernel(const long long* __restrict__ ei,
    const float* __restrict__ ELOG, const float* __restrict__ DEN,
    const float* __restrict__ H, float* __restrict__ AGG) {
  const int lane = threadIdx.x & 31;
  const int e    = blockIdx.x * 8 + (threadIdx.x >> 5);
  int s, d;
  if (e < N_EDGES) { s = (int)ei[e]; d = (int)ei[N_EDGES + e]; }
  else             { s = d = e - N_EDGES; }
  float alpha = ELOG[e] / (DEN[d] + 1e-16f);
  float4 hv = ((const float4*)(H + (size_t)s * HID))[lane];
  float* out = AGG + (size_t)d * HID + lane * 4;
  atomicAdd(out + 0, hv.x * alpha);
  atomicAdd(out + 1, hv.y * alpha);
  atomicAdd(out + 2, hv.z * alpha);
  atomicAdd(out + 3, hv.w * alpha);
}

// Y = relu(agg + bias); grid exact (N*HID/256)
__global__ __launch_bounds__(256) void finish_kernel(const float* __restrict__ AGG,
    const float* __restrict__ b, float* __restrict__ Y) {
  int i = blockIdx.x * blockDim.x + threadIdx.x;
  float v = AGG[i] + b[i & 127];
  Y[i] = v > 0.f ? v : 0.f;
}

// Head: out[n,c] = H[n,:] . Wlin[:,c] + blin[c]   (Wlin tiny -> cache-resident)
__global__ __launch_bounds__(256) void linear_kernel(const float* __restrict__ H,
    const float* __restrict__ Wl, const float* __restrict__ bl,
    float* __restrict__ out) {
  int i = blockIdx.x * blockDim.x + threadIdx.x;
  if (i >= N_NODES * NCLS) return;
  int node = i / NCLS, c = i % NCLS;
  const float* hr = H + (size_t)node * HID;
  float acc = bl[c];
#pragma unroll 8
  for (int k = 0; k < HID; ++k) acc += hr[k] * Wl[k * NCLS + c];
  out[i] = acc;
}

// ---------------------------------------------------------------------------
static void run_layer(const float* Xin, const float* W,
                      const float* asv, const float* adv, const float* bv,
                      const long long* ei,
                      float* Hbuf, float* AG, float* AS, float* AD,
                      unsigned* MX, float* DEN, float* ELOG,
                      unsigned short* WThi, unsigned short* WTlo,
                      hipStream_t stream) {
  const int ET = N_EDGES + N_NODES;
  wconv_kernel     <<<(HID * HID) / 256,       256, 0, stream>>>(W, WThi, WTlo);
  gemm_kernel      <<<N_NODES / ROWS_PER_BLOCK, 256, 0, stream>>>(Xin, WThi, WTlo, Hbuf);
  alpha_init_kernel<<<N_NODES / 8,             256, 0, stream>>>(Hbuf, asv, adv, AS, AD, MX, DEN, AG);
  edge_max_kernel  <<<(ET + 255) / 256,        256, 0, stream>>>(ei, AS, AD, ELOG, MX);
  edge_exp_kernel  <<<(ET + 255) / 256,        256, 0, stream>>>(ei, ELOG, MX, DEN);
  aggregate_kernel <<<ET / 8,                  256, 0, stream>>>(ei, ELOG, DEN, Hbuf, AG);
  finish_kernel    <<<(N_NODES * HID) / 256,   256, 0, stream>>>(AG, bv, Hbuf);
}

extern "C" void kernel_launch(void* const* d_in, const int* in_sizes, int n_in,
                              void* d_out, int out_size, void* d_ws, size_t ws_size,
                              hipStream_t stream) {
  const float*     x   = (const float*)d_in[0];
  const long long* ei  = (const long long*)d_in[1];
  // d_in[2] = batch (unused: single graph)
  const float* W1  = (const float*)d_in[3];
  const float* as1 = (const float*)d_in[4];
  const float* ad1 = (const float*)d_in[5];
  const float* b1  = (const float*)d_in[6];
  const float* W2  = (const float*)d_in[7];
  const float* as2 = (const float*)d_in[8];
  const float* ad2 = (const float*)d_in[9];
  const float* b2  = (const float*)d_in[10];
  const float* Wl  = (const float*)d_in[11];
  const float* bl  = (const float*)d_in[12];
  float* out = (float*)d_out;

  char* p = (char*)d_ws;
  auto carve = [&](size_t bytes) -> char* {
    char* q = p;
    p += (bytes + 255) & ~(size_t)255;
    return q;
  };
  float*          HA   = (float*)carve((size_t)N_NODES * HID * 4);
  float*          HB   = (float*)carve((size_t)N_NODES * HID * 4);
  float*          AG   = (float*)carve((size_t)N_NODES * HID * 4);
  float*          AS   = (float*)carve((size_t)N_NODES * 4);
  float*          AD   = (float*)carve((size_t)N_NODES * 4);
  unsigned*       MX   = (unsigned*)carve((size_t)N_NODES * 4);
  float*          DEN  = (float*)carve((size_t)N_NODES * 4);
  float*          ELOG = (float*)carve((size_t)(N_EDGES + N_NODES) * 4);
  unsigned short* WThi = (unsigned short*)carve((size_t)HID * HID * 2);
  unsigned short* WTlo = (unsigned short*)carve((size_t)HID * HID * 2);

  // Layer 1: x -> HA
  run_layer(x,  W1, as1, ad1, b1, ei, HA, AG, AS, AD, MX, DEN, ELOG, WThi, WTlo, stream);
  // Layer 2: HA -> HB
  run_layer(HA, W2, as2, ad2, b2, ei, HB, AG, AS, AD, MX, DEN, ELOG, WThi, WTlo, stream);
  // Head
  linear_kernel<<<(N_NODES * NCLS + 255) / 256, 256, 0, stream>>>(HB, Wl, bl, out);

  (void)in_sizes; (void)n_in; (void)out_size; (void)ws_size;
}
